// SpatialProcessor_10557029614168
// MI455X (gfx1250) — matrix operand order repaired and checked by reading, not compiled, not run on hardware
//
#include <hip/hip_runtime.h>

typedef __attribute__((ext_vector_type(16))) _Float16 v16h;
typedef __attribute__((ext_vector_type(8)))  _Float16 v8h;
typedef __attribute__((ext_vector_type(8)))  float    v8f;

#define NB 2
#define NN 512
#define DIN 32
#define NH 4
#define DD 64          // per-head dim (both layers)
#define CW 256         // H*D channel width (both layers)
#define NEG_SLOPE 0.2f
#define LN_EPS 1e-5f

// ---------------------------------------------------------------------------
// mask[i,j] = (e_i . e_j != 0) || (i==j)   -> additive bias 0 / -1e30
// (rsqrt normalization never changes the zero pattern, so skip it)
// ---------------------------------------------------------------------------
__global__ void mask_kernel(const float* __restrict__ emb, float* __restrict__ maskbias) {
    __shared__ float s_ei[DD];
    int i = blockIdx.x, tid = threadIdx.x;
    if (tid < DD) s_ei[tid] = emb[i * DD + tid];
    __syncthreads();
    for (int j = tid; j < NN; j += 256) {
        const float* ej = emb + (long)j * DD;
        float dot = 0.f;
#pragma unroll 8
        for (int d = 0; d < DD; ++d) dot += s_ei[d] * ej[d];
        maskbias[(long)i * NN + j] = (dot != 0.f || j == i) ? 0.f : -1e30f;
    }
}

// ---------------------------------------------------------------------------
// Elementwise f32 -> f16 (activation pre-convert, hoists cvt out of GEMM).
// ---------------------------------------------------------------------------
__global__ void cvt_f16(const float* __restrict__ in, _Float16* __restrict__ out, long n) {
    long idx = (long)blockIdx.x * 256 + threadIdx.x;
    if (idx < n) out[idx] = (_Float16)in[idx];
}

// ---------------------------------------------------------------------------
// Weight transpose+convert: W[K][CW] f32 -> WT[CW][K] f16 (K-contiguous).
// ---------------------------------------------------------------------------
__global__ void transpose_w_f16(const float* __restrict__ W, _Float16* __restrict__ WT, int K) {
    long idx = (long)blockIdx.x * 256 + threadIdx.x;
    long total = (long)K * CW;
    if (idx < total) {
        int n = (int)(idx / K), k = (int)(idx % K);
        WT[idx] = (_Float16)W[(long)k * CW + n];
    }
}

// ---------------------------------------------------------------------------
// xl transpose+convert: XL[(b*NN+j)*CW + h*DD + d] f32
//                    -> XLT[((b*NH+h)*DD + d)*NN + j] f16 (j-contiguous)
// ---------------------------------------------------------------------------
__global__ void transpose_xl_f16(const float* __restrict__ XL, _Float16* __restrict__ XLT) {
    long idx = (long)blockIdx.x * 256 + threadIdx.x;   // over NB*NH*DD*NN
    int j = (int)(idx % NN);
    long t = idx / NN;
    int d = (int)(t % DD);
    long t2 = t / DD;
    int h = (int)(t2 % NH);
    int b = (int)(t2 / NH);
    XLT[idx] = (_Float16)XL[((long)(b * NN + j)) * CW + h * DD + d];
}

// ---------------------------------------------------------------------------
// WMMA GEMM: C[z] = A[z] (MxK, f16 row-major) @ Bt[z]^T (Bt: NcxK f16,
// K-contiguous) + bias.  One wave -> 16x64 C tile: one A fragment feeds
// 4 unrolled v_wmma_f32_16x16x32_f16 per K step.  All fragment loads are
// contiguous 16B runs per ISA 7.12.2 -> global_load_b128.
// ---------------------------------------------------------------------------
__global__ void gemm_wmma(const _Float16* __restrict__ A, int lda, long sA,
                          const _Float16* __restrict__ Bt, int ldbt, long sBt,
                          float* __restrict__ C, int ldc, long sCb, long sCh,
                          int K, const float* __restrict__ bias, int Hdiv) {
    const int z  = blockIdx.z;
    const int bz = z / Hdiv, hz = z % Hdiv;
    const _Float16* Az = A  + (long)z * sA;
    const _Float16* Bz = Bt + (long)z * sBt;
    float*          Cz = C  + (long)bz * sCb + (long)hz * sCh;

    const int lane  = threadIdx.x;
    const int tm    = blockIdx.y * 16;
    const int nbase = blockIdx.x * 64;
    const int hi    = (lane >= 16);
    const int r15   = lane & 15;

    v8f c[4];
#pragma unroll
    for (int u = 0; u < 4; ++u) {
        const float binit = bias ? bias[nbase + u * 16 + r15] : 0.f;
#pragma unroll
        for (int r = 0; r < 8; ++r) c[u][r] = binit;
    }

    const int aRow = tm + r15;
    const int kbA  = hi ? 8 : 0;    // A: lanes>=16 hold K 8..15 / 24..31
    const int kbB  = hi ? 8 : 0;    // B: lanes>=16 hold K 8..15 / 24..31

    for (int k0 = 0; k0 < K; k0 += 32) {
        // A fragment: halves [kbA..kbA+7] and [kbA+16..kbA+23] -> 2x b128
        const v8h* pa = (const v8h*)(Az + (long)aRow * lda + k0 + kbA);
        const v8h alo = pa[0];
        const v8h ahi = pa[2];
        v16h a;
#pragma unroll
        for (int i = 0; i < 8; ++i) { a[i] = alo[i]; a[8 + i] = ahi[i]; }

        if (k0 + 32 < K)
            __builtin_prefetch(Az + (long)aRow * lda + k0 + 32 + kbA, 0, 3);

#pragma unroll
        for (int u = 0; u < 4; ++u) {
            // B fragment: halves [kbB..kbB+7] and [kbB+16..kbB+23] of column (nbase+16u+r15) -> 2x b128
            const v8h* pb = (const v8h*)(Bz + (long)(nbase + u * 16 + r15) * ldbt + k0 + kbB);
            const v8h blo = pb[0];
            const v8h bhi = pb[2];
            v16h b;
#pragma unroll
            for (int i = 0; i < 8; ++i) { b[i] = blo[i]; b[8 + i] = bhi[i]; }

            c[u] = __builtin_amdgcn_wmma_f32_16x16x32_f16(
                       /*neg_a=*/false, a, /*neg_b=*/false, b,
                       /*c_mod=*/(short)0, c[u], /*reuse_a=*/false, /*reuse_b=*/false);
        }
    }

#pragma unroll
    for (int u = 0; u < 4; ++u) {
        const int col = nbase + u * 16 + r15;
#pragma unroll
        for (int r = 0; r < 8; ++r) {
            const int row = tm + r + (hi ? 8 : 0);
            Cz[(long)row * ldc + col] = c[u][r];
        }
    }
}

// ---------------------------------------------------------------------------
// Fused GATv2 logits + masked softmax for one (b,h,i) row.
// e_j = att_h . leaky_relu(xl_j + xr_i) + maskbias[i,j]; alpha written as f16.
// leaky_relu inside the contraction prevents a matmul form -> fp32 VALU.
// ---------------------------------------------------------------------------
__global__ void attn_softmax(const float* __restrict__ XL, const float* __restrict__ XR,
                             const float* __restrict__ att, const float* __restrict__ maskbias,
                             _Float16* __restrict__ alpha) {
    const int i = blockIdx.x, h = blockIdx.y, b = blockIdx.z, tid = threadIdx.x;
    __shared__ float s_xr[DD], s_att[DD], red[256];
    if (tid < DD) {
        s_xr[tid]  = XR[((long)(b * NN + i)) * CW + h * DD + tid];
        s_att[tid] = att[h * DD + tid];
    }
    __syncthreads();

    float e0[2];
#pragma unroll
    for (int t = 0; t < 2; ++t) {
        const int j = tid + t * 256;
        const float4* xlj = (const float4*)(XL + ((long)(b * NN + j)) * CW + h * DD);
        float acc = 0.f;
#pragma unroll 4
        for (int q = 0; q < DD / 4; ++q) {
            float4 v = xlj[q];
            float z;
            z = v.x + s_xr[4*q+0]; z = z > 0.f ? z : NEG_SLOPE * z; acc += s_att[4*q+0] * z;
            z = v.y + s_xr[4*q+1]; z = z > 0.f ? z : NEG_SLOPE * z; acc += s_att[4*q+1] * z;
            z = v.z + s_xr[4*q+2]; z = z > 0.f ? z : NEG_SLOPE * z; acc += s_att[4*q+2] * z;
            z = v.w + s_xr[4*q+3]; z = z > 0.f ? z : NEG_SLOPE * z; acc += s_att[4*q+3] * z;
        }
        e0[t] = acc + maskbias[(long)i * NN + j];
    }

    red[tid] = fmaxf(e0[0], e0[1]);
    __syncthreads();
    for (int s = 128; s > 0; s >>= 1) {
        if (tid < s) red[tid] = fmaxf(red[tid], red[tid + s]);
        __syncthreads();
    }
    const float m = red[0];
    __syncthreads();

    const float p0 = __expf(e0[0] - m), p1 = __expf(e0[1] - m);
    red[tid] = p0 + p1;
    __syncthreads();
    for (int s = 128; s > 0; s >>= 1) {
        if (tid < s) red[tid] += red[tid + s];
        __syncthreads();
    }
    const float rinv = 1.f / red[0];

    _Float16* arow = alpha + (((long)(b * NH + h)) * NN + i) * NN;
    arow[tid]       = (_Float16)(p0 * rinv);
    arow[tid + 256] = (_Float16)(p1 * rinv);
}

// ---------------------------------------------------------------------------
// y = Y_row + badd;  LayerNorm(y)*g + be;  optional ReLU.  One block per row.
// ---------------------------------------------------------------------------
__global__ void bias_ln(const float* __restrict__ Y, const float* __restrict__ badd,
                        const float* __restrict__ g, const float* __restrict__ be,
                        float* __restrict__ Out, int relu) {
    const long row = blockIdx.x;
    const int tid = threadIdx.x;
    __shared__ float red[256];

    const float y = Y[row * CW + tid] + badd[tid];
    red[tid] = y;
    __syncthreads();
    for (int s = 128; s > 0; s >>= 1) { if (tid < s) red[tid] += red[tid + s]; __syncthreads(); }
    const float mu = red[0] * (1.f / CW);
    __syncthreads();

    const float d = y - mu;
    red[tid] = d * d;
    __syncthreads();
    for (int s = 128; s > 0; s >>= 1) { if (tid < s) red[tid] += red[tid + s]; __syncthreads(); }
    const float var = red[0] * (1.f / CW);

    float o = d * rsqrtf(var + LN_EPS) * g[tid] + be[tid];
    if (relu) o = fmaxf(o, 0.f);
    Out[row * CW + tid] = o;
}

// ---------------------------------------------------------------------------
extern "C" void kernel_launch(void* const* d_in, const int* in_sizes, int n_in,
                              void* d_out, int out_size, void* d_ws, size_t ws_size,
                              hipStream_t stream) {
    const float* x     = (const float*)d_in[0];
    const float* emb   = (const float*)d_in[1];
    const float* w1l   = (const float*)d_in[2];
    const float* b1l   = (const float*)d_in[3];
    const float* w1r   = (const float*)d_in[4];
    const float* b1r   = (const float*)d_in[5];
    const float* att1  = (const float*)d_in[6];
    const float* bias1 = (const float*)d_in[7];
    const float* g1    = (const float*)d_in[8];
    const float* be1   = (const float*)d_in[9];
    const float* w2l   = (const float*)d_in[10];
    const float* b2l   = (const float*)d_in[11];
    const float* w2r   = (const float*)d_in[12];
    const float* b2r   = (const float*)d_in[13];
    const float* att2  = (const float*)d_in[14];
    const float* bias2 = (const float*)d_in[15];
    const float* g2    = (const float*)d_in[16];
    const float* be2   = (const float*)d_in[17];

    char* ws = (char*)d_ws;
    float*    maskbias = (float*)ws;    ws += (size_t)NN * NN * 4;            // 1 MB
    float*    xl       = (float*)ws;    ws += (size_t)NB * NN * CW * 4;       // 1 MB
    float*    xr       = (float*)ws;    ws += (size_t)NB * NN * CW * 4;       // 1 MB
    float*    agg      = (float*)ws;    ws += (size_t)NB * NN * CW * 4;       // 1 MB
    float*    hbuf     = (float*)ws;    ws += (size_t)NB * NN * CW * 4;       // 1 MB
    _Float16* alpha    = (_Float16*)ws; ws += (size_t)NB * NH * NN * NN * 2;  // 4 MB
    _Float16* xh16     = (_Float16*)ws; ws += (size_t)NB * NN * CW * 2;       // 512 KB
    _Float16* xlT      = (_Float16*)ws; ws += (size_t)NB * NH * DD * NN * 2;  // 512 KB
    _Float16* w1lT     = (_Float16*)ws; ws += (size_t)DIN * CW * 2;           // 16 KB
    _Float16* w1rT     = (_Float16*)ws; ws += (size_t)DIN * CW * 2;           // 16 KB
    _Float16* w2lT     = (_Float16*)ws; ws += (size_t)CW * CW * 2;            // 128 KB
    _Float16* w2rT     = (_Float16*)ws; ws += (size_t)CW * CW * 2;            // 128 KB

    mask_kernel<<<dim3(NN), dim3(256), 0, stream>>>(emb, maskbias);

    const dim3 gProj(CW / 64, (NB * NN) / 16, 1);   // 4 x 64 tiles (16x64 each)
    const dim3 gAgg(DD / 64, NN / 16, NB * NH);     // 1 x 32 x 8 tiles
    const long n_x1 = (long)NB * NN * DIN;
    const long n_hc = (long)NB * NN * CW;
    const long n_xt = (long)NB * NH * DD * NN;

    // ---- layer 1 ----
    cvt_f16<<<dim3((n_x1 + 255) / 256), 256, 0, stream>>>(x, xh16, n_x1);
    transpose_w_f16<<<dim3((DIN * CW + 255) / 256), 256, 0, stream>>>(w1l, w1lT, DIN);
    transpose_w_f16<<<dim3((DIN * CW + 255) / 256), 256, 0, stream>>>(w1r, w1rT, DIN);
    gemm_wmma<<<gProj, 32, 0, stream>>>(xh16, DIN, 0, w1lT, DIN, 0,
                                        xl, CW, 0, 0, DIN, b1l, 1);
    gemm_wmma<<<gProj, 32, 0, stream>>>(xh16, DIN, 0, w1rT, DIN, 0,
                                        xr, CW, 0, 0, DIN, b1r, 1);
    attn_softmax<<<dim3(NN, NH, NB), 256, 0, stream>>>(xl, xr, att1, maskbias, alpha);
    transpose_xl_f16<<<dim3((n_xt + 255) / 256), 256, 0, stream>>>(xl, xlT);
    gemm_wmma<<<gAgg, 32, 0, stream>>>(alpha, NN, (long)NN * NN, xlT, NN, (long)DD * NN,
                                       agg, CW, (long)NN * CW, DD, NN, nullptr, NH);
    bias_ln<<<dim3(NB * NN), 256, 0, stream>>>(agg, bias1, g1, be1, hbuf, /*relu=*/1);

    // ---- layer 2 ----
    cvt_f16<<<dim3((n_hc + 255) / 256), 256, 0, stream>>>(hbuf, xh16, n_hc);
    transpose_w_f16<<<dim3((CW * CW + 255) / 256), 256, 0, stream>>>(w2l, w2lT, CW);
    transpose_w_f16<<<dim3((CW * CW + 255) / 256), 256, 0, stream>>>(w2r, w2rT, CW);
    gemm_wmma<<<gProj, 32, 0, stream>>>(xh16, CW, 0, w2lT, CW, 0,
                                        xl, CW, 0, 0, CW, b2l, 1);
    gemm_wmma<<<gProj, 32, 0, stream>>>(xh16, CW, 0, w2rT, CW, 0,
                                        xr, CW, 0, 0, CW, b2r, 1);
    attn_softmax<<<dim3(NN, NH, NB), 256, 0, stream>>>(xl, xr, att2, maskbias, alpha);
    transpose_xl_f16<<<dim3((n_xt + 255) / 256), 256, 0, stream>>>(xl, xlT);
    gemm_wmma<<<gAgg, 32, 0, stream>>>(alpha, NN, (long)NN * NN, xlT, NN, (long)DD * NN,
                                       agg, CW, (long)NN * CW, DD, NN, nullptr, NH);
    bias_ln<<<dim3(NB * NN), 256, 0, stream>>>(agg, bias2, g2, be2, (float*)d_out, /*relu=*/0);

    (void)in_sizes; (void)n_in; (void)out_size; (void)ws_size;
}
